// CSNet_12472585028065
// MI455X (gfx1250) — compile-verified, matrix operand order checked by reading, not run on hardware
//
#include <hip/hip_runtime.h>
#include <hip/hip_bf16.h>
#include <stddef.h>

typedef _Float16 half_t;
typedef __attribute__((ext_vector_type(16))) _Float16 v16h;
typedef __attribute__((ext_vector_type(8)))  _Float16 v8h;
typedef __attribute__((ext_vector_type(8)))  float    v8f;
typedef __attribute__((ext_vector_type(4)))  float    v4f;

#define IMG 512
#define PH  514
#define PW  514
#define NB  4
#define NC  64
#define NPIX ((size_t)NB * IMG * IMG)          // 1,048,576
#define NELE ((size_t)NB * IMG * IMG * NC)     // 67,108,864
#define WPACK_HALVES (9 * 2 * 64 * 32)         // 36,864 per conv

// ---------------------------------------------------------------------------
// Weight repack: OIHW f32 [64,64,3,3] -> f16 [tap][kc][co][ci32]
// ---------------------------------------------------------------------------
__global__ __launch_bounds__(256) void pack_w_kernel(const float* __restrict__ w,
                                                     half_t* __restrict__ dst) {
  int i = blockIdx.x * 256 + threadIdx.x;           // 36864 total
  if (i >= WPACK_HALVES) return;
  int ci32 = i & 31;
  int t    = i >> 5;
  int co   = t & 63;
  int tk   = t >> 6;                                 // tap*2 + kc
  int kc   = tk & 1;
  int tap  = tk >> 1;
  int ci   = kc * 32 + ci32;
  dst[i] = (half_t)w[(co * 64 + ci) * 9 + tap];
}

// ---------------------------------------------------------------------------
// conv1 (1x1, 64->256) fused with depth_to_space(bs=16) -> padded f32 [4][518][518]
// ---------------------------------------------------------------------------
__global__ __launch_bounds__(256) void conv1_d2s_kernel(const float* __restrict__ x,
                                                        const float* __restrict__ w,
                                                        const float* __restrict__ bias,
                                                        float* __restrict__ y0p) {
  int idx = blockIdx.x * 256 + threadIdx.x;          // 4*512*512
  int xx = idx & 511;
  int t  = idx >> 9;
  int yy = t & 511;
  int b  = t >> 9;
  int i = yy >> 4, k1 = yy & 15, j = xx >> 4, k2 = xx & 15;
  int c = k1 * 16 + k2;                              // source channel
  float acc = bias[c];
  const float* xb = x + (size_t)b * 64 * 1024 + i * 32 + j;
  const float* wc = w + c * 64;
#pragma unroll 8
  for (int ci = 0; ci < 64; ++ci) acc += xb[(size_t)ci * 1024] * wc[ci];
  y0p[((size_t)(b * 518) + yy + 3) * 518 + xx + 3] = acc;
}

// ---------------------------------------------------------------------------
// blk1: 7x7 conv 1->64 + PReLU -> padded NHWC f16
// ---------------------------------------------------------------------------
__global__ __launch_bounds__(256) void blk1_kernel(const float* __restrict__ y0p,
                                                   const float* __restrict__ w,
                                                   const float* __restrict__ bias,
                                                   const float* __restrict__ alpha,
                                                   half_t* __restrict__ b1) {
  int idx = blockIdx.x * 256 + threadIdx.x;          // 4*512*512
  int xx = idx & 511;
  int t  = idx >> 9;
  int yy = t & 511;
  int b  = t >> 9;
  float in[49];
  const float* base = y0p + ((size_t)(b * 518) + yy) * 518 + xx;
#pragma unroll
  for (int dy = 0; dy < 7; ++dy)
#pragma unroll
    for (int dx = 0; dx < 7; ++dx) in[dy * 7 + dx] = base[dy * 518 + dx];
  float a = alpha[0];
  half_t* op = b1 + (((size_t)(b * PH + yy + 1)) * PW + xx + 1) * 64;
  for (int co = 0; co < 64; ++co) {
    float acc = bias[co];
    const float* wc = w + co * 49;
#pragma unroll
    for (int tt = 0; tt < 49; ++tt) acc += in[tt] * wc[tt];
    acc = acc >= 0.f ? acc : a * acc;
    op[co] = (half_t)acc;
  }
}

// ---------------------------------------------------------------------------
// Main 3x3 64->64 conv, implicit GEMM on WMMA f16.
//   in  : padded NHWC f16 [4][514][514][64]
//   wpk : packed f16 [9][2][64][32] (staged in LDS)
//   out : NHWC f32 [4][512][512][64] (+bias)
// Wave: 2 pixel-rows x 16 pixels (M) x 64 co (4 N-tiles). Block = 8 waves.
// ---------------------------------------------------------------------------
__global__ __launch_bounds__(256) void conv3x3_wmma_kernel(
    const half_t* __restrict__ inp, const half_t* __restrict__ wpk,
    const float* __restrict__ bias, float* __restrict__ out) {
  extern __shared__ __align__(16) char smem_raw[];
  half_t* lw = (half_t*)smem_raw;                    // 36864 halves = 73728 B

  // cooperative weight stage: 4608 x 16B
  {
    const v4f* src = (const v4f*)wpk;
    v4f* dst = (v4f*)lw;
#pragma unroll
    for (int k = 0; k < 18; ++k) dst[k * 256 + threadIdx.x] = src[k * 256 + threadIdx.x];
  }
  __syncthreads();

  const int lane = threadIdx.x & 31;
  const int wv   = threadIdx.x >> 5;                 // 0..7
  const int l16  = lane & 15;
  const int h    = lane >> 4;                        // K-subgroup / M-subgroup select
  const int x0 = blockIdx.x * 16;
  const int y0 = blockIdx.y * 16;
  const int b  = blockIdx.z;
  const int yb = y0 + wv * 2;

  // A-fragment per-lane base: pixel m=l16 at (yb, x0+l16), channel sub-offset h*8
  const half_t* abase =
      inp + (((size_t)(b * PH + (yb + 1)) * PW + (x0 + l16 + 1)) * 64 + h * 8);
  // B-fragment per-lane base offset in LDS halves (n = l16)
  const int bbase = l16 * 32 + h * 8;

  v8f acc[2][4];
#pragma unroll
  for (int r = 0; r < 2; ++r)
#pragma unroll
    for (int nt = 0; nt < 4; ++nt) acc[r][nt] = v8f{0.f, 0.f, 0.f, 0.f, 0.f, 0.f, 0.f, 0.f};

#pragma unroll
  for (int tap = 0; tap < 9; ++tap) {
    const int dy = tap / 3 - 1;
    const int dx = tap % 3 - 1;
#pragma unroll
    for (int kc = 0; kc < 2; ++kc) {
      v16h bf[4];
#pragma unroll
      for (int nt = 0; nt < 4; ++nt) {
        int off = ((tap * 2 + kc) * 64 + nt * 16) * 32 + bbase;
        v8h blo = *(const v8h*)(lw + off);
        v8h bhi = *(const v8h*)(lw + off + 16);
        bf[nt] = __builtin_shufflevector(blo, bhi, 0, 1, 2, 3, 4, 5, 6, 7, 8, 9, 10,
                                         11, 12, 13, 14, 15);
      }
#pragma unroll
      for (int r = 0; r < 2; ++r) {
        const half_t* ap = abase + ((ptrdiff_t)(r + dy) * PW + dx) * 64 + kc * 32;
        v8h alo = *(const v8h*)ap;
        v8h ahi = *(const v8h*)(ap + 16);
        v16h af = __builtin_shufflevector(alo, ahi, 0, 1, 2, 3, 4, 5, 6, 7, 8, 9, 10,
                                          11, 12, 13, 14, 15);
#pragma unroll
        for (int nt = 0; nt < 4; ++nt)
          acc[r][nt] = __builtin_amdgcn_wmma_f32_16x16x32_f16(
              false, af, false, bf[nt], (short)0, acc[r][nt], false, false);
      }
    }
  }

  // store D (+bias). D layout: lane = n + 16*hm ; VGPR j -> M = hm*8 + j
#pragma unroll
  for (int r = 0; r < 2; ++r) {
    int y = yb + r;
#pragma unroll
    for (int nt = 0; nt < 4; ++nt) {
      int co = nt * 16 + l16;
      float bv = bias[co];
      float* op = out + (((size_t)(b * IMG + y) * IMG + x0 + h * 8) * 64 + co);
#pragma unroll
      for (int j = 0; j < 8; ++j) op[(size_t)j * 64] = acc[r][nt][j] + bv;
    }
  }
}

// ---------------------------------------------------------------------------
// Per-channel sum / sumsq reduction over NHWC f32 (for training-mode BN)
// ---------------------------------------------------------------------------
__global__ __launch_bounds__(256) void stats_kernel(const float* __restrict__ u,
                                                    float* __restrict__ sums,
                                                    unsigned n) {
  __shared__ float ls[128];
  if (threadIdx.x < 128) ls[threadIdx.x] = 0.f;
  __syncthreads();
  int c = threadIdx.x & 63;
  float s = 0.f, s2 = 0.f;
  size_t stride = (size_t)gridDim.x * 256;           // multiple of 64 -> c fixed
  for (size_t i = (size_t)blockIdx.x * 256 + threadIdx.x; i < n; i += stride) {
    float v = u[i];
    s += v;
    s2 += v * v;
  }
  atomicAdd(&ls[c], s);
  atomicAdd(&ls[64 + c], s2);
  __syncthreads();
  if (threadIdx.x < 128) atomicAdd(&sums[threadIdx.x], ls[threadIdx.x]);
}

// ---------------------------------------------------------------------------
// Fused elementwise: mode 0 = BN+PReLU ; mode 1 = BN + residual add ;
//                    mode 2 = PReLU + add b1 (no BN)
// reads f32 conv out, writes f16 into padded NHWC buffer
// ---------------------------------------------------------------------------
__global__ __launch_bounds__(256) void ew_kernel(const float* __restrict__ u,
                                                 const float* __restrict__ sums,
                                                 const float* __restrict__ gamma,
                                                 const float* __restrict__ beta,
                                                 const float* __restrict__ alpha,
                                                 const half_t* __restrict__ res,
                                                 half_t* __restrict__ outp, int mode) {
  size_t idx = (size_t)blockIdx.x * 256 + threadIdx.x;   // exactly NELE threads
  int c = (int)(idx & 63);
  size_t pix = idx >> 6;
  int x = (int)(pix & 511);
  size_t t = pix >> 9;
  int y = (int)(t & 511);
  int b = (int)(t >> 9);
  float v = u[idx];
  if (mode < 2) {
    const float invN = 1.f / (float)NPIX;
    float m   = sums[c] * invN;
    float var = sums[64 + c] * invN - m * m;
    float sc  = gamma[c] * rsqrtf(var + 1e-5f);
    v = (v - m) * sc + beta[c];
  }
  if (mode != 1) {                                   // PReLU for modes 0 and 2
    float a = alpha[0];
    v = v >= 0.f ? v : a * v;
  }
  size_t po = (((size_t)(b * PH + y + 1)) * PW + (x + 1)) * 64 + c;
  if (mode >= 1) v += (float)res[po];                // residual / +b1
  outp[po] = (half_t)v;
}

// ---------------------------------------------------------------------------
// blk8: 3x3 conv 64->1 from padded f16 -> f32 NCHW output
// ---------------------------------------------------------------------------
__global__ __launch_bounds__(256) void blk8_kernel(const half_t* __restrict__ s,
                                                   const float* __restrict__ w,
                                                   const float* __restrict__ bias,
                                                   float* __restrict__ out) {
  int idx = blockIdx.x * 256 + threadIdx.x;          // 4*512*512
  int xx = idx & 511;
  int t  = idx >> 9;
  int yy = t & 511;
  int b  = t >> 9;
  float acc = bias[0];
  const half_t* base = s + (((size_t)(b * PH + yy)) * PW + xx) * 64;
#pragma unroll
  for (int dy = 0; dy < 3; ++dy)
#pragma unroll
    for (int dx = 0; dx < 3; ++dx) {
      const half_t* p = base + ((size_t)dy * PW + dx) * 64;
      int tt = dy * 3 + dx;
#pragma unroll 8
      for (int ci = 0; ci < 64; ++ci) acc += (float)p[ci] * w[ci * 9 + tt];
    }
  out[idx] = acc;
}

// ---------------------------------------------------------------------------
// host-side orchestration
// ---------------------------------------------------------------------------
extern "C" void kernel_launch(void* const* d_in, const int* in_sizes, int n_in,
                              void* d_out, int out_size, void* d_ws, size_t ws_size,
                              hipStream_t stream) {
  (void)in_sizes; (void)n_in; (void)out_size; (void)ws_size;

  const size_t PADB = (size_t)NB * PH * PW * 64 * sizeof(half_t);   // 135.3 MB
  const size_t UB   = NELE * sizeof(float);                          // 256 MB
  const size_t Y0B  = (size_t)NB * 518 * 518 * sizeof(float);        // 4.3 MB

  char* ws = (char*)d_ws;
  size_t off = 0;
  auto carve = [&](size_t bytes) {
    void* p = ws + off;
    off += (bytes + 255) & ~(size_t)255;
    return p;
  };
  half_t* B1  = (half_t*)carve(PADB);
  half_t* Hb  = (half_t*)carve(PADB);
  half_t* Tb  = (half_t*)carve(PADB);
  float*  U   = (float*)carve(UB);
  float*  y0p = (float*)carve(Y0B);
  half_t* wpk = (half_t*)carve((size_t)11 * WPACK_HALVES * sizeof(half_t));
  float*  sums = (float*)carve(128 * sizeof(float));

  // zero padded buffers (borders must be 0; interiors get overwritten)
  hipMemsetAsync(B1, 0, PADB, stream);
  hipMemsetAsync(Hb, 0, PADB, stream);
  hipMemsetAsync(Tb, 0, PADB, stream);
  hipMemsetAsync(y0p, 0, Y0B, stream);

  // input indices (setup_inputs dict order)
  const float* x_in   = (const float*)d_in[0];
  const float* c1_w   = (const float*)d_in[1];
  const float* c1_b   = (const float*)d_in[2];
  const float* b1_w   = (const float*)d_in[3];
  const float* b1_b   = (const float*)d_in[4];
  const float* b1_al  = (const float*)d_in[5];
  // rb k (k=0..4): base 6+9k : c1_w,c1_b,bn1_g,bn1_b,alpha,c2_w,c2_b,bn2_g,bn2_b
  const float* b7_w  = (const float*)d_in[51];
  const float* b7_b  = (const float*)d_in[52];
  const float* b7_al = (const float*)d_in[53];
  const float* b8_w  = (const float*)d_in[54];
  const float* b8_b  = (const float*)d_in[55];

  // pack all 3x3 conv weights: j = 2k -> rb(k)_c1, 2k+1 -> rb(k)_c2, 10 -> blk7
  const int packBlocks = (WPACK_HALVES + 255) / 256;
  for (int k = 0; k < 5; ++k) {
    int pb = 6 + 9 * k;
    pack_w_kernel<<<packBlocks, 256, 0, stream>>>((const float*)d_in[pb + 0],
                                                  wpk + (size_t)(2 * k) * WPACK_HALVES);
    pack_w_kernel<<<packBlocks, 256, 0, stream>>>((const float*)d_in[pb + 5],
                                                  wpk + (size_t)(2 * k + 1) * WPACK_HALVES);
  }
  pack_w_kernel<<<packBlocks, 256, 0, stream>>>(b7_w, wpk + (size_t)10 * WPACK_HALVES);

  const int pixBlocks = (int)(NPIX / 256);           // 4096
  conv1_d2s_kernel<<<pixBlocks, 256, 0, stream>>>(x_in, c1_w, c1_b, y0p);
  blk1_kernel<<<pixBlocks, 256, 0, stream>>>(y0p, b1_w, b1_b, b1_al, B1);

  const dim3 cgrid(IMG / 16, IMG / 16, NB);
  const size_t ldsB = (size_t)WPACK_HALVES * sizeof(half_t);  // 73728
  const int eweBlocks = (int)(NELE / 256);           // 262144

  const half_t* hin = B1;
  for (int k = 0; k < 5; ++k) {
    int pb = 6 + 9 * k;
    // conv1 of block -> BN -> PReLU -> Tb
    conv3x3_wmma_kernel<<<cgrid, 256, ldsB, stream>>>(
        hin, wpk + (size_t)(2 * k) * WPACK_HALVES, (const float*)d_in[pb + 1], U);
    hipMemsetAsync(sums, 0, 128 * sizeof(float), stream);
    stats_kernel<<<1024, 256, 0, stream>>>(U, sums, (unsigned)NELE);
    ew_kernel<<<eweBlocks, 256, 0, stream>>>(U, sums, (const float*)d_in[pb + 2],
                                             (const float*)d_in[pb + 3],
                                             (const float*)d_in[pb + 4], B1, Tb, 0);
    // conv2 of block -> BN -> + residual -> Hb
    conv3x3_wmma_kernel<<<cgrid, 256, ldsB, stream>>>(
        Tb, wpk + (size_t)(2 * k + 1) * WPACK_HALVES, (const float*)d_in[pb + 6], U);
    hipMemsetAsync(sums, 0, 128 * sizeof(float), stream);
    stats_kernel<<<1024, 256, 0, stream>>>(U, sums, (unsigned)NELE);
    ew_kernel<<<eweBlocks, 256, 0, stream>>>(U, sums, (const float*)d_in[pb + 7],
                                             (const float*)d_in[pb + 8],
                                             (const float*)d_in[pb + 4], hin, Hb, 1);
    hin = Hb;
  }

  // blk7: conv -> prelu -> + b1 -> Tb
  conv3x3_wmma_kernel<<<cgrid, 256, ldsB, stream>>>(
      Hb, wpk + (size_t)10 * WPACK_HALVES, b7_b, U);
  ew_kernel<<<eweBlocks, 256, 0, stream>>>(U, sums, b7_b, b7_b, b7_al, B1, Tb, 2);

  // blk8: 3x3 64->1
  blk8_kernel<<<pixBlocks, 256, 0, stream>>>(Tb, b8_w, b8_b, (float*)d_out);
}
